// WaveNet_89172110999688
// MI455X (gfx1250) — matrix operand order repaired and checked by reading, not compiled, not run on hardware
//
#include <hip/hip_runtime.h>
#include <math.h>
#include <stdint.h>

typedef _Float16 h16;
typedef _Float16 v16h __attribute__((ext_vector_type(16)));
typedef float    v8f  __attribute__((ext_vector_type(8)));

#define DEVINL __device__ __forceinline__

// ---- WMMA fragment gathers (wave32, v_wmma_f32_16x16x32_f16 layouts) ----
// A (16x32 f16, MxK): lane<16 -> M=lane, K in {0..7}U{16..23}; lane>=16 -> M=lane-16, K+8.
DEVINL v16h frag_A(const h16* p, int laneHalf) {
  union { v16h h; uint32_t u[8]; } r;
  const uint32_t* q = (const uint32_t*)p;
  int o = laneHalf * 4;
  r.u[0]=q[o+0]; r.u[1]=q[o+1]; r.u[2]=q[o+2]; r.u[3]=q[o+3];
  r.u[4]=q[o+8]; r.u[5]=q[o+9]; r.u[6]=q[o+10]; r.u[7]=q[o+11];
  return r.h;
}
// B (32x16 f16, KxN): lane<16 -> N=lane, K in {0..7}U{16..23}; lane>=16 -> N=lane-16, K+8.
// p = WT + n*Kpad + kb  (weights stored transposed [Nout][Kpad], row-major).
DEVINL v16h frag_B(const h16* p, int laneHalf) {
  union { v16h h; uint32_t u[8]; } r;
  const uint32_t* q = (const uint32_t*)p + laneHalf * 4;
#pragma unroll
  for (int v = 0; v < 8; ++v) r.u[v] = q[v + (v >> 2) * 4];
  return r.h;
}
DEVINL v8f wmma_f16(v16h a, v16h b, v8f c) {
  return __builtin_amdgcn_wmma_f32_16x16x32_f16(false, a, false, b, (short)0, c, false, false);
}
DEVINL float sigm(float x) { return 1.f / (1.f + expf(-x)); }

// ---------------- conversion / weight repack ----------------
__global__ void k_f32h(const float* __restrict__ s, h16* __restrict__ d, long n) {
  long i = (long)blockIdx.x * blockDim.x + threadIdx.x;
  if (i < n) d[i] = (h16)s[i];
}
// src shape (taps, Cin, Cout) row-major f32 -> dst f16 [NoutPad][Kpad],
// dst[n][tap*CinBlk + ci], zero padded.
__global__ void k_pack(const float* __restrict__ src, h16* __restrict__ dst,
                       int taps, int Cin, int Cout, int CinBlk, int Kpad, int NoutPad) {
  int idx = blockIdx.x * blockDim.x + threadIdx.x;
  if (idx >= NoutPad * Kpad) return;
  int n = idx / Kpad, k = idx % Kpad;
  int tap = k / CinBlk, ci = k % CinBlk;
  float v = 0.f;
  if (n < Cout && tap < taps && ci < Cin) v = src[((tap * Cin) + ci) * Cout + n];
  dst[idx] = (h16)v;
}

// ---------------- causal conv: GEMM M=16383/batch, K=512 (rows t,t+1 contiguous), N=256 ----
// One wave computes a 16x64 tile (4 N-tiles) so the A stream is read once per 4 N-tiles.
__global__ void k_causal(const h16* __restrict__ xh, const h16* __restrict__ WT,
                         const float* __restrict__ bias, h16* __restrict__ out) {
  int lane = threadIdx.x & 31, laneHalf = lane >> 4, lidx = lane & 15;
  int gw = blockIdx.x * 8 + (threadIdx.x >> 5);
  int ng = gw & 3; gw >>= 2;                   // N-group of 64 columns
  int mtile = gw & 1023; int b = gw >> 10;
  int t0 = mtile * 16;
  const h16* arow = xh + ((size_t)b * 16384 + t0 + lidx) * 256;
  __builtin_prefetch(arow + 16 * 256, 0, 1);   // next M-tile's A rows -> global_prefetch
  const h16* br = WT + (size_t)(ng * 64 + lidx) * 512;
  v8f acc[4] = {{}, {}, {}, {}};
#pragma unroll 2
  for (int kb = 0; kb < 512; kb += 32) {
    v16h a = frag_A(arow + kb, laneHalf);
#pragma unroll
    for (int j = 0; j < 4; ++j)
      acc[j] = wmma_f16(a, frag_B(br + (size_t)j * 16 * 512 + kb, laneHalf), acc[j]);
  }
#pragma unroll
  for (int j = 0; j < 4; ++j) {
    int n = ng * 64 + j * 16 + lidx;
    float bs = bias[n];
#pragma unroll
    for (int v = 0; v < 8; ++v) {
      int m = t0 + v + 8 * laneHalf;
      if (m < 16383) out[((size_t)b * 16384 + m) * 256 + n] = (h16)(acc[j][v] + bs);
    }
  }
}

// ---------------- gated block 0: K=512 (contig), 256 -> 24(pad32), fused h=tanh(f)*sig(g), 1x1 ----
__global__ void k_gated0(const h16* __restrict__ in, const h16* __restrict__ gT,
                         const h16* __restrict__ fT, const h16* __restrict__ sT,
                         const float* __restrict__ gb, const float* __restrict__ fb,
                         const float* __restrict__ sb, h16* __restrict__ out) {
  __shared__ h16 hbuf[8][16 * 32];
  int wave = threadIdx.x >> 5;
  int lane = threadIdx.x & 31, laneHalf = lane >> 4, lidx = lane & 15;
  int gw = blockIdx.x * 8 + wave;
  int mtile = gw & 1023; int b = gw >> 10;
  int t0 = mtile * 16;
  const h16* arow = in + ((size_t)b * 16384 + t0 + lidx) * 256;
  __builtin_prefetch(arow + 16 * 256, 0, 1);
  const h16* g0 = gT + (size_t)lidx * 512;
  const h16* g1 = gT + (size_t)(16 + lidx) * 512;
  const h16* f0 = fT + (size_t)lidx * 512;
  const h16* f1 = fT + (size_t)(16 + lidx) * 512;
  v8f ag0 = {}, ag1 = {}, af0 = {}, af1 = {};
  for (int kb = 0; kb < 512; kb += 32) {
    v16h a = frag_A(arow + kb, laneHalf);
    ag0 = wmma_f16(a, frag_B(g0 + kb, laneHalf), ag0);
    ag1 = wmma_f16(a, frag_B(g1 + kb, laneHalf), ag1);
    af0 = wmma_f16(a, frag_B(f0 + kb, laneHalf), af0);
    af1 = wmma_f16(a, frag_B(f1 + kb, laneHalf), af1);
  }
  int n0 = lidx, n1 = 16 + lidx;
  float gb0 = (n0 < 24) ? gb[n0] : 0.f, gb1 = (n1 < 24) ? gb[n1] : 0.f;
  float fb0 = (n0 < 24) ? fb[n0] : 0.f, fb1 = (n1 < 24) ? fb[n1] : 0.f;
#pragma unroll
  for (int v = 0; v < 8; ++v) {
    int m = v + 8 * laneHalf;
    hbuf[wave][m * 32 + n0] = (h16)(tanhf(af0[v] + fb0) * sigm(ag0[v] + gb0));
    hbuf[wave][m * 32 + n1] = (h16)(tanhf(af1[v] + fb1) * sigm(ag1[v] + gb1));
  }
  __syncthreads();
  v16h ah = frag_A(&hbuf[wave][lidx * 32], laneHalf);
  v8f o0 = {}, o1 = {};
  o0 = wmma_f16(ah, frag_B(sT + (size_t)lidx * 32, laneHalf), o0);
  o1 = wmma_f16(ah, frag_B(sT + (size_t)(16 + lidx) * 32, laneHalf), o1);
  float sb0 = (n0 < 24) ? sb[n0] : 0.f, sb1 = (n1 < 24) ? sb[n1] : 0.f;
#pragma unroll
  for (int v = 0; v < 8; ++v) {
    int m = t0 + v + 8 * laneHalf;
    if (m < 16382) {
      size_t ro = ((size_t)b * 16384 + m) * 32;
      out[ro + n0] = (h16)(o0[v] + sb0);
      out[ro + n1] = (h16)(o1[v] + sb1);
    }
  }
}

// ---------------- dilated gated block: taps t and t+d, 24(pad32)->24(pad32), fused ----
__global__ void k_gated_dil(const h16* __restrict__ in, h16* __restrict__ out,
                            const h16* __restrict__ gT, const h16* __restrict__ fT,
                            const h16* __restrict__ sT, const float* __restrict__ gb,
                            const float* __restrict__ fb, const float* __restrict__ sb,
                            int d, int Lout, int Mtiles) {
  __shared__ h16 hbuf[8][16 * 32];
  int wave = threadIdx.x >> 5;
  int lane = threadIdx.x & 31, laneHalf = lane >> 4, lidx = lane & 15;
  int gw = blockIdx.x * 8 + wave;
  int mtile = gw % Mtiles; int b = gw / Mtiles;
  int t0 = mtile * 16;
  const h16* a0 = in + ((size_t)b * 16384 + t0 + lidx) * 32;      // tap 0
  const h16* a1 = a0 + (size_t)d * 32;                            // tap 1
  v16h A0 = frag_A(a0, laneHalf), A1 = frag_A(a1, laneHalf);
  const h16* gr0 = gT + (size_t)lidx * 64;
  const h16* gr1 = gT + (size_t)(16 + lidx) * 64;
  const h16* fr0 = fT + (size_t)lidx * 64;
  const h16* fr1 = fT + (size_t)(16 + lidx) * 64;
  v8f ag0 = {}, ag1 = {}, af0 = {}, af1 = {};
  ag0 = wmma_f16(A0, frag_B(gr0, laneHalf), ag0);
  ag0 = wmma_f16(A1, frag_B(gr0 + 32, laneHalf), ag0);
  ag1 = wmma_f16(A0, frag_B(gr1, laneHalf), ag1);
  ag1 = wmma_f16(A1, frag_B(gr1 + 32, laneHalf), ag1);
  af0 = wmma_f16(A0, frag_B(fr0, laneHalf), af0);
  af0 = wmma_f16(A1, frag_B(fr0 + 32, laneHalf), af0);
  af1 = wmma_f16(A0, frag_B(fr1, laneHalf), af1);
  af1 = wmma_f16(A1, frag_B(fr1 + 32, laneHalf), af1);
  int n0 = lidx, n1 = 16 + lidx;
  float gb0 = (n0 < 24) ? gb[n0] : 0.f, gb1 = (n1 < 24) ? gb[n1] : 0.f;
  float fb0 = (n0 < 24) ? fb[n0] : 0.f, fb1 = (n1 < 24) ? fb[n1] : 0.f;
#pragma unroll
  for (int v = 0; v < 8; ++v) {
    int m = v + 8 * laneHalf;
    hbuf[wave][m * 32 + n0] = (h16)(tanhf(af0[v] + fb0) * sigm(ag0[v] + gb0));
    hbuf[wave][m * 32 + n1] = (h16)(tanhf(af1[v] + fb1) * sigm(ag1[v] + gb1));
  }
  __syncthreads();
  v16h ah = frag_A(&hbuf[wave][lidx * 32], laneHalf);
  v8f o0 = {}, o1 = {};
  o0 = wmma_f16(ah, frag_B(sT + (size_t)lidx * 32, laneHalf), o0);
  o1 = wmma_f16(ah, frag_B(sT + (size_t)(16 + lidx) * 32, laneHalf), o1);
  float sb0 = (n0 < 24) ? sb[n0] : 0.f, sb1 = (n1 < 24) ? sb[n1] : 0.f;
#pragma unroll
  for (int v = 0; v < 8; ++v) {
    int m = t0 + v + 8 * laneHalf;
    if (m < Lout) {
      size_t ro = ((size_t)b * 16384 + m) * 32;
      out[ro + n0] = (h16)(o0[v] + sb0);
      out[ro + n1] = (h16)(o1[v] + sb1);
    }
  }
}

// ---------------- res: (prev7[t+256] + out8[t]) @ resW -> relu -> f16 (B,15360,128) ----
// One wave handles all 8 N-tiles; A fragment gathered once.
__global__ void k_res(const h16* __restrict__ prev, const h16* __restrict__ cur,
                      const h16* __restrict__ rT, const float* __restrict__ rb,
                      h16* __restrict__ ares) {
  int lane = threadIdx.x & 31, laneHalf = lane >> 4, lidx = lane & 15;
  int gw = blockIdx.x * 8 + (threadIdx.x >> 5);
  int mtile = gw % 960; int b = gw / 960;
  int t0 = mtile * 16;
  const h16* pa = prev + ((size_t)b * 16384 + t0 + lidx + 256) * 32;
  const h16* pb = cur + ((size_t)b * 16384 + t0 + lidx) * 32;
  v16h a = frag_A(pa, laneHalf) + frag_A(pb, laneHalf);
#pragma unroll
  for (int nt = 0; nt < 8; ++nt) {
    v8f acc = {};
    acc = wmma_f16(a, frag_B(rT + (size_t)(nt * 16 + lidx) * 32, laneHalf), acc);
    int n = nt * 16 + lidx;
    float bs = rb[n];
#pragma unroll
    for (int v = 0; v < 8; ++v) {
      int m = t0 + v + 8 * laneHalf;
      ares[((size_t)b * 15360 + m) * 128 + n] = (h16)fmaxf(acc[v] + bs, 0.f);
    }
  }
}

// ---------------- generic 1x1 GEMM over (B,15360,Kdim) -> N=256; relu->f16 or raw->f32 ----
// One wave computes a 16x64 tile (4 N-tiles), A fragments read once per K-step.
__global__ void k_gemm(const h16* __restrict__ A, const h16* __restrict__ WT,
                       const float* __restrict__ bias, int Kdim, int relu,
                       h16* __restrict__ outh, float* __restrict__ outf) {
  int lane = threadIdx.x & 31, laneHalf = lane >> 4, lidx = lane & 15;
  int gw = blockIdx.x * 8 + (threadIdx.x >> 5);
  int ng = gw & 3; gw >>= 2;
  int mtile = gw % 960; int b = gw / 960;
  const h16* arow = A + ((size_t)b * 15360 + mtile * 16 + lidx) * (size_t)Kdim;
  __builtin_prefetch(arow + 16 * (size_t)Kdim, 0, 1);
  const h16* br = WT + (size_t)(ng * 64 + lidx) * (size_t)Kdim;
  v8f acc[4] = {{}, {}, {}, {}};
  for (int kb = 0; kb < Kdim; kb += 32) {
    v16h a = frag_A(arow + kb, laneHalf);
#pragma unroll
    for (int j = 0; j < 4; ++j)
      acc[j] = wmma_f16(a, frag_B(br + (size_t)j * 16 * Kdim + kb, laneHalf), acc[j]);
  }
#pragma unroll
  for (int j = 0; j < 4; ++j) {
    int n = ng * 64 + j * 16 + lidx;
    float bs = bias[n];
#pragma unroll
    for (int v = 0; v < 8; ++v) {
      int m = mtile * 16 + v + 8 * laneHalf;
      size_t o = ((size_t)b * 15360 + m) * 256 + n;
      float val = acc[j][v] + bs;
      if (relu) outh[o] = (h16)fmaxf(val, 0.f);
      else      outf[o] = val;
    }
  }
}

// ---------------- softmax over 256 channels, one wave per row ----------------
__global__ void k_softmax(const float* __restrict__ logits, float* __restrict__ out) {
  int lane = threadIdx.x & 31;
  int row = blockIdx.x * 8 + (threadIdx.x >> 5);
  const float* p = logits + (size_t)row * 256;
  float v[8], mx = -3.4e38f;
#pragma unroll
  for (int i = 0; i < 8; ++i) { v[i] = p[lane + 32 * i]; mx = fmaxf(mx, v[i]); }
  for (int off = 16; off; off >>= 1) mx = fmaxf(mx, __shfl_xor(mx, off, 32));
  float sum = 0.f;
#pragma unroll
  for (int i = 0; i < 8; ++i) { v[i] = expf(v[i] - mx); sum += v[i]; }
  for (int off = 16; off; off >>= 1) sum += __shfl_xor(sum, off, 32);
  float inv = 1.f / sum;
  float* q = out + (size_t)row * 256;
#pragma unroll
  for (int i = 0; i < 8; ++i) q[lane + 32 * i] = v[i] * inv;
}

extern "C" void kernel_launch(void* const* d_in, const int* in_sizes, int n_in,
                              void* d_out, int out_size, void* d_ws, size_t ws_size,
                              hipStream_t stream) {
  (void)in_sizes; (void)n_in; (void)out_size; (void)ws_size;
  const float* x        = (const float*)d_in[0];
  const float* causal_W = (const float*)d_in[1];
  const float* causal_b = (const float*)d_in[2];
  const float* gW0      = (const float*)d_in[3];
  const float* gb0      = (const float*)d_in[4];
  const float* fW0      = (const float*)d_in[5];
  const float* fb0      = (const float*)d_in[6];
  const float* sW0      = (const float*)d_in[7];
  const float* sb0      = (const float*)d_in[8];
  const float* gate_W   = (const float*)d_in[9];
  const float* gate_b   = (const float*)d_in[10];
  const float* filter_W = (const float*)d_in[11];
  const float* filter_b = (const float*)d_in[12];
  const float* scale_W  = (const float*)d_in[13];
  const float* scale_b  = (const float*)d_in[14];
  const float* res_W    = (const float*)d_in[15];
  const float* res_b    = (const float*)d_in[16];
  const float* f1_W     = (const float*)d_in[17];
  const float* f1_b     = (const float*)d_in[18];
  const float* f2_W     = (const float*)d_in[19];
  const float* f2_b     = (const float*)d_in[20];

  char* ws = (char*)d_ws;
  size_t off = 0;
  auto alloc = [&](size_t bytes) { size_t o = off; off += (bytes + 255) & ~(size_t)255; return o; };
  const size_t bigH = (size_t)8 * 16384 * 256 + 8192;   // halves, +slack rows
  const size_t smallH = (size_t)8 * 16384 * 32 + 8192;
  size_t o_xh  = alloc(bigH * 2);
  size_t o_o1  = alloc(bigH * 2);
  size_t o_aA  = alloc(smallH * 2);
  size_t o_aB  = alloc(smallH * 2);
  size_t o_ar  = alloc((size_t)8 * 15360 * 128 * 2 + 16384);
  size_t o_a1  = alloc((size_t)8 * 15360 * 256 * 2 + 16384);
  size_t o_WcT = alloc(131072 * 2);
  size_t o_g0  = alloc(16384 * 2);
  size_t o_f0  = alloc(16384 * 2);
  size_t o_s0  = alloc(1024 * 2);
  size_t o_gd  = alloc(9 * 2048 * 2);
  size_t o_fd  = alloc(9 * 2048 * 2);
  size_t o_sd  = alloc(9 * 1024 * 2);
  size_t o_rT  = alloc(4096 * 2);
  size_t o_f1T = alloc(32768 * 2);
  size_t o_f2T = alloc(65536 * 2);

  h16* xh   = (h16*)(ws + o_xh);
  h16* o1h  = (h16*)(ws + o_o1);
  h16* actA = (h16*)(ws + o_aA);
  h16* actB = (h16*)(ws + o_aB);
  h16* ares = (h16*)(ws + o_ar);
  h16* a1   = (h16*)(ws + o_a1);
  h16* WcT  = (h16*)(ws + o_WcT);
  h16* g0T  = (h16*)(ws + o_g0);
  h16* f0T  = (h16*)(ws + o_f0);
  h16* s0T  = (h16*)(ws + o_s0);
  h16* gdT  = (h16*)(ws + o_gd);
  h16* fdT  = (h16*)(ws + o_fd);
  h16* sdT  = (h16*)(ws + o_sd);
  h16* rT   = (h16*)(ws + o_rT);
  h16* f1T  = (h16*)(ws + o_f1T);
  h16* f2T  = (h16*)(ws + o_f2T);
  // fp32 logits (126 MB) reuse the dead xh+o1h region (134 MB) at kernel-6 time.
  float* logits = (float*)(ws + o_xh);

  // ---- weight repack + input convert ----
  k_pack<<<512, 256, 0, stream>>>(causal_W, WcT, 2, 256, 256, 256, 512, 256);
  k_pack<<<64, 256, 0, stream>>>(gW0, g0T, 2, 256, 24, 256, 512, 32);
  k_pack<<<64, 256, 0, stream>>>(fW0, f0T, 2, 256, 24, 256, 512, 32);
  k_pack<<<4, 256, 0, stream>>>(sW0, s0T, 1, 24, 24, 32, 32, 32);
  for (int i = 0; i < 9; ++i) {
    k_pack<<<8, 256, 0, stream>>>(gate_W + (size_t)i * 1152, gdT + (size_t)i * 2048, 2, 24, 24, 32, 64, 32);
    k_pack<<<8, 256, 0, stream>>>(filter_W + (size_t)i * 1152, fdT + (size_t)i * 2048, 2, 24, 24, 32, 64, 32);
    k_pack<<<4, 256, 0, stream>>>(scale_W + (size_t)i * 576, sdT + (size_t)i * 1024, 1, 24, 24, 32, 32, 32);
  }
  k_pack<<<16, 256, 0, stream>>>(res_W + (size_t)8 * 24 * 128, rT, 1, 24, 128, 32, 32, 128);
  k_pack<<<128, 256, 0, stream>>>(f1_W, f1T, 1, 128, 256, 128, 128, 256);
  k_pack<<<256, 256, 0, stream>>>(f2_W, f2T, 1, 256, 256, 256, 256, 256);
  k_f32h<<<131072, 256, 0, stream>>>(x, xh, (long)8 * 16384 * 256);

  // ---- network ----
  k_causal<<<4096, 256, 0, stream>>>(xh, WcT, causal_b, o1h);           // 8*1024*4 waves
  k_gated0<<<1024, 256, 0, stream>>>(o1h, g0T, f0T, s0T, gb0, fb0, sb0, actA);

  h16* bufs[2] = {actA, actB};
  int Lin = 16382;
  for (int i = 0; i < 9; ++i) {
    int d = 2 << i;                 // 2,4,...,512
    int Lout = Lin - d;
    int Mtiles = (Lout + 15) / 16;
    k_gated_dil<<<Mtiles, 256, 0, stream>>>(bufs[i % 2], bufs[(i + 1) % 2],
        gdT + (size_t)i * 2048, fdT + (size_t)i * 2048, sdT + (size_t)i * 1024,
        gate_b + (size_t)i * 24, filter_b + (size_t)i * 24, scale_b + (size_t)i * 24,
        d, Lout, Mtiles);
    Lin = Lout;
  }
  // layer 7 output sits in actA, layer 8 output in actB; trimmed = prev7[t+256]
  k_res<<<960, 256, 0, stream>>>(actA, actB, rT, res_b + 8 * 128, ares);
  k_gemm<<<3840, 256, 0, stream>>>(ares, f1T, f1_b, 128, 1, a1, nullptr);   // relu -> f16
  k_gemm<<<3840, 256, 0, stream>>>(a1, f2T, f2_b, 256, 0, nullptr, logits); // raw -> f32
  k_softmax<<<15360, 256, 0, stream>>>(logits, (float*)d_out);
}
